// GBSWT5GroupedQueryAttention_55473797595726
// MI455X (gfx1250) — compile-verified, matrix operand order checked by reading, not run on hardware
//
#include <hip/hip_runtime.h>
#include <hip/hip_bf16.h>

// ---------------------------------------------------------------------------
// Types for CDNA5 WMMA (wave32, 16x16x32 bf16 -> f32)
// ---------------------------------------------------------------------------
typedef __bf16 bf16_t;
typedef __attribute__((ext_vector_type(16))) __bf16 v16bf;
typedef __attribute__((ext_vector_type(8)))  __bf16 v8bf;
typedef __attribute__((ext_vector_type(8)))  float  v8f;

#define S_LEN   2048
#define DMODEL  1024
#define INNER   1024
#define HKV     4
#define GRP     4
#define DK      64
#define BATCH   2

// 16-bit A/B fragment (ISA 7.12.2): per lane, elements are two contiguous
// 8-element runs within a 32-long K row: k = 8h..8h+7 and k = 16+8h..23+8h.
// With the tile stored row-major along K, that is two aligned 16-byte loads.
__device__ __forceinline__ v16bf frag_from_row(const bf16_t* row, int half) {
  v8bf lo = *reinterpret_cast<const v8bf*>(row + 8 * half);
  v8bf hi = *reinterpret_cast<const v8bf*>(row + 16 + 8 * half);
  return __builtin_shufflevector(lo, hi, 0, 1, 2, 3, 4, 5, 6, 7,
                                 8, 9, 10, 11, 12, 13, 14, 15);
}

// ---------------------------------------------------------------------------
// f32 -> bf16 conversion
// ---------------------------------------------------------------------------
__global__ void cvt_f32_bf16_kernel(const float* __restrict__ src,
                                    bf16_t* __restrict__ dst, int n) {
  int i = blockIdx.x * blockDim.x + threadIdx.x;
  if (i < n) dst[i] = (bf16_t)src[i];
}

// ---------------------------------------------------------------------------
// Tiled bf16 WMMA GEMM: C[M,N] = A[M,K] * B[K,N]
// Block = 256 threads (8 waves). Block tile 128(M) x 64(N), k-step 32.
// A staged row-major [128][32]; B staged TRANSPOSED [64 n][32 k] so both
// fragment types load as 2x ds_load_b128 per lane.
// ---------------------------------------------------------------------------
template <typename OUT_T>
__global__ __launch_bounds__(256) void gemm_bf16_wmma_kernel(
    const bf16_t* __restrict__ A, const bf16_t* __restrict__ B,
    OUT_T* __restrict__ C, int M, int N, int K) {
  __shared__ __align__(16) bf16_t As[128 * 32];
  __shared__ __align__(16) bf16_t BsT[64 * 32];

  const int tid  = threadIdx.x;
  const int wave = tid >> 5;
  const int lane = tid & 31;
  const int half = lane >> 4;
  const int nl   = lane & 15;
  const int m0   = blockIdx.x * 128;
  const int n0   = blockIdx.y * 64;

  v8f acc[4];
#pragma unroll
  for (int nt = 0; nt < 4; ++nt)
#pragma unroll
    for (int j = 0; j < 8; ++j) acc[nt][j] = 0.0f;

  for (int k0 = 0; k0 < K; k0 += 32) {
    __syncthreads();
    // Stage A tile: 128 rows x 32 bf16 = 2048 dwords, 8 per thread (coalesced)
#pragma unroll
    for (int i = 0; i < 8; ++i) {
      int idx = tid + i * 256;          // 0..2047
      int row = idx >> 4;               // 16 dwords per row
      int c2  = idx & 15;
      reinterpret_cast<unsigned int*>(As)[idx] =
          reinterpret_cast<const unsigned int*>(A + (size_t)(m0 + row) * K + k0)[c2];
    }
    // Stage B tile transposed: 512 2x2 bf16 blocks, 2 per thread.
    // Coalesced b32 global reads from two adjacent K rows, register 2x2
    // transpose, paired ds_store_b32 into BsT[n][k].
#pragma unroll
    for (int i = 0; i < 2; ++i) {
      int bid = tid + i * 256;          // 0..511
      int nb  = (bid & 31) * 2;         // n pair
      int kb  = (bid >> 5) * 2;         // k pair
      unsigned int r0 = reinterpret_cast<const unsigned int*>(
          B + (size_t)(k0 + kb) * N + n0)[nb >> 1];
      unsigned int r1 = reinterpret_cast<const unsigned int*>(
          B + (size_t)(k0 + kb + 1) * N + n0)[nb >> 1];
      unsigned int w0 = (r0 & 0xFFFFu) | (r1 << 16);
      unsigned int w1 = (r0 >> 16) | (r1 & 0xFFFF0000u);
      unsigned int* BT = reinterpret_cast<unsigned int*>(BsT);
      BT[nb * 16 + (kb >> 1)]       = w0;
      BT[(nb + 1) * 16 + (kb >> 1)] = w1;
    }
    __syncthreads();

    v16bf a = frag_from_row(As + (wave * 16 + nl) * 32, half);
#pragma unroll
    for (int nt = 0; nt < 4; ++nt) {
      v16bf b = frag_from_row(BsT + (nt * 16 + nl) * 32, half);
      acc[nt] = __builtin_amdgcn_wmma_f32_16x16x32_bf16(
          false, a, false, b, (short)0, acc[nt], false, false);
    }
  }

  // D layout: lane nl = col, vgpr j = row (+8 for upper half-wave)
#pragma unroll
  for (int nt = 0; nt < 4; ++nt)
#pragma unroll
    for (int j = 0; j < 8; ++j) {
      int row = m0 + wave * 16 + j + 8 * half;
      int col = n0 + nt * 16 + nl;
      C[(size_t)row * N + col] = (OUT_T)acc[nt][j];
    }
}

// ---------------------------------------------------------------------------
// Qsum[b,hk,s,d] = sum_g Q[b,s,(g*4+hk)*64+d] ; Kmean = mean_g K[...]
// ---------------------------------------------------------------------------
__global__ void reduce_heads_kernel(const bf16_t* __restrict__ Qb,
                                    const bf16_t* __restrict__ Kb,
                                    bf16_t* __restrict__ Qsum,
                                    bf16_t* __restrict__ Kmean) {
  int i = blockIdx.x * blockDim.x + threadIdx.x;
  if (i >= BATCH * HKV * S_LEN * DK) return;
  int d  = i & 63;
  int s  = (i >> 6) & (S_LEN - 1);
  int hk = (i >> 17) & 3;
  int b  = i >> 19;
  float qs = 0.f, ks = 0.f;
  size_t base = ((size_t)(b * S_LEN + s)) * INNER;
#pragma unroll
  for (int g = 0; g < GRP; ++g) {
    int hh = g * HKV + hk;
    qs += (float)Qb[base + hh * DK + d];
    ks += (float)Kb[base + hh * DK + d];
  }
  size_t o = ((size_t)((b * HKV + hk) * S_LEN + s)) * DK + d;
  Qsum[o]  = (bf16_t)qs;
  Kmean[o] = (bf16_t)(ks * 0.25f);
}

// ---------------------------------------------------------------------------
// biasmean[bkt*4+hk] = mean_g rel_bias[bkt*16 + g*4 + hk]
// ---------------------------------------------------------------------------
__global__ void bias_prep_kernel(const float* __restrict__ rel,
                                 float* __restrict__ biasmean) {
  int t = threadIdx.x;
  if (t < 128) {
    int bkt = t >> 2, hk = t & 3;
    float s = 0.f;
#pragma unroll
    for (int g = 0; g < GRP; ++g) s += rel[bkt * 16 + g * HKV + hk];
    biasmean[t] = s * 0.25f;
  }
}

// ---------------------------------------------------------------------------
// Fused flash-attention (scores + T5 bias + online softmax + P@V).
// grid.x = S/128 query blocks, grid.y = B*HKV. 8 waves, wave = one 16-row
// query tile. Keys processed in chunks of 32.
//  - K chunk staged [key][dim] (fragment-friendly: dim is the K axis)
//  - V chunks staged TRANSPOSED [dim][key] (key is the K axis for P@V)
//  - P re-layout through wave-private LDS, guarded by s_wait_dscnt
// ---------------------------------------------------------------------------
__global__ __launch_bounds__(256) void flash_attn_kernel(
    const bf16_t* __restrict__ Qsum,   // [B*HKV, S, 64]
    const bf16_t* __restrict__ Kmean,  // [B*HKV, S, 64]
    const bf16_t* __restrict__ V,      // [B, S, 1024]
    const float*  __restrict__ biasmean,  // [32*4]
    bf16_t* __restrict__ Ctx)          // [B, S, 1024]
{
  __shared__ __align__(16) bf16_t Kt[32 * 64];        // [key][dim]
  __shared__ __align__(16) bf16_t VtT[GRP][64 * 32];  // [dim][key]
  __shared__ __align__(16) bf16_t Pb[8][16 * 32];     // [qrow][key]
  __shared__ float bm[128];

  const int tid  = threadIdx.x;
  const int wave = tid >> 5;
  const int lane = tid & 31;
  const int half = lane >> 4;
  const int nl   = lane & 15;
  const int bh   = blockIdx.y;          // b*HKV + hk
  const int b    = bh >> 2;
  const int hk   = bh & 3;
  const int qbase = blockIdx.x * 128 + wave * 16;

  if (tid < 128) bm[tid] = biasmean[tid];

  // Q A-fragments from global (rows are 128B, so 2x global_load_b128 each)
  const bf16_t* Qrow = Qsum + ((size_t)bh * S_LEN) * DK;
  v16bf a0 = frag_from_row(Qrow + (size_t)(qbase + nl) * DK, half);
  v16bf a1 = frag_from_row(Qrow + (size_t)(qbase + nl) * DK + 32, half);

  v8f O[GRP][4];
#pragma unroll
  for (int g = 0; g < GRP; ++g)
#pragma unroll
    for (int nt = 0; nt < 4; ++nt)
#pragma unroll
      for (int j = 0; j < 8; ++j) O[g][nt][j] = 0.0f;

  float mrow[8], lrow[8];
#pragma unroll
  for (int j = 0; j < 8; ++j) { mrow[j] = -1e30f; lrow[j] = 0.0f; }

  const bf16_t* Kg = Kmean + ((size_t)bh * S_LEN) * DK;
  const bf16_t* Vg = V + ((size_t)b * S_LEN) * INNER;

  for (int s0 = 0; s0 < S_LEN; s0 += 32) {
    __syncthreads();
    // Stage K chunk [key][dim]: 1024 dwords, coalesced
#pragma unroll
    for (int i = 0; i < 4; ++i) {
      int idx = tid + i * 256;
      int key = idx >> 5, c2 = idx & 31;
      reinterpret_cast<unsigned int*>(Kt)[idx] =
          reinterpret_cast<const unsigned int*>(Kg + (size_t)(s0 + key) * DK)[c2];
    }
    // Stage V chunks transposed [dim][key] via register 2x2 transpose
#pragma unroll
    for (int g = 0; g < GRP; ++g)
#pragma unroll
      for (int i = 0; i < 2; ++i) {
        int bid = tid + i * 256;        // 0..511
        int db  = (bid & 31) * 2;       // dim pair
        int kb  = (bid >> 5) * 2;       // key pair
        const bf16_t* vb = Vg + (g * HKV + hk) * DK;
        unsigned int r0 = reinterpret_cast<const unsigned int*>(
            vb + (size_t)(s0 + kb) * INNER)[db >> 1];
        unsigned int r1 = reinterpret_cast<const unsigned int*>(
            vb + (size_t)(s0 + kb + 1) * INNER)[db >> 1];
        unsigned int w0 = (r0 & 0xFFFFu) | (r1 << 16);
        unsigned int w1 = (r0 >> 16) | (r1 & 0xFFFF0000u);
        unsigned int* VT = reinterpret_cast<unsigned int*>(VtT[g]);
        VT[db * 16 + (kb >> 1)]       = w0;
        VT[(db + 1) * 16 + (kb >> 1)] = w1;
      }
    if (s0 + 32 < S_LEN)  // global_prefetch_b8 for the next K chunk
      __builtin_prefetch(Kg + (size_t)(s0 + 32) * DK, 0, 1);
    __syncthreads();

    // Scores: two 16-key tiles, each = 2 WMMAs over the 64-d contraction.
    // B fragment rows are Kt[key][...]: dim runs contiguous -> 2x b128 loads.
    v8f sc[2];
#pragma unroll
    for (int nt = 0; nt < 2; ++nt) {
      const bf16_t* krow = Kt + (nt * 16 + nl) * 64;
      v16bf b0 = frag_from_row(krow, half);
      v16bf b1 = frag_from_row(krow + 32, half);
      v8f z;
#pragma unroll
      for (int j = 0; j < 8; ++j) z[j] = 0.0f;
      v8f c0 = __builtin_amdgcn_wmma_f32_16x16x32_bf16(
          false, a0, false, b0, (short)0, z, false, false);
      sc[nt] = __builtin_amdgcn_wmma_f32_16x16x32_bf16(
          false, a1, false, b1, (short)0, c0, false, false);
    }

    // T5 relative-position bias (bidirectional, 32 buckets, max_dist 128)
    float tmax[8];
#pragma unroll
    for (int j = 0; j < 8; ++j) {
      int row = qbase + j + 8 * half;   // query position n
      float mj = -1e30f;
#pragma unroll
      for (int nt = 0; nt < 2; ++nt) {
        int scol = s0 + nt * 16 + nl;   // key position s
        int rel  = scol - row;          // mem - ctx
        int bkt  = (rel > 0) ? 16 : 0;
        int a    = rel < 0 ? -rel : rel;
        int add;
        if (a < 8) {
          add = a;
        } else {
          int li = 8 + (int)(__logf((float)a * 0.125f) * (8.0f / 2.7725887f));
          add = li < 15 ? li : 15;
        }
        float val = sc[nt][j] + bm[(bkt + add) * 4 + hk];
        sc[nt][j] = val;
        mj = fmaxf(mj, val);
      }
      tmax[j] = mj;
    }
    // Row max across the 16 key-lanes of each half-wave
#pragma unroll
    for (int j = 0; j < 8; ++j) {
      float v = tmax[j];
      v = fmaxf(v, __shfl_xor(v, 1));
      v = fmaxf(v, __shfl_xor(v, 2));
      v = fmaxf(v, __shfl_xor(v, 4));
      v = fmaxf(v, __shfl_xor(v, 8));
      tmax[j] = v;
    }
    // Online softmax update
    float scale[8];
#pragma unroll
    for (int j = 0; j < 8; ++j) {
      float nm = fmaxf(mrow[j], tmax[j]);
      scale[j] = __expf(mrow[j] - nm);
      mrow[j]  = nm;
    }
#pragma unroll
    for (int g = 0; g < GRP; ++g)
#pragma unroll
      for (int nt = 0; nt < 4; ++nt)
#pragma unroll
        for (int j = 0; j < 8; ++j) O[g][nt][j] *= scale[j];
#pragma unroll
    for (int j = 0; j < 8; ++j) {
      float p0 = __expf(sc[0][j] - mrow[j]);
      float p1 = __expf(sc[1][j] - mrow[j]);
      sc[0][j] = p0; sc[1][j] = p1;
      float r = p0 + p1;
      r += __shfl_xor(r, 1);
      r += __shfl_xor(r, 2);
      r += __shfl_xor(r, 4);
      r += __shfl_xor(r, 8);
      lrow[j] = lrow[j] * scale[j] + r;
    }

    // Re-layout P (C-layout -> A-layout) through wave-private LDS
#pragma unroll
    for (int nt = 0; nt < 2; ++nt)
#pragma unroll
      for (int j = 0; j < 8; ++j) {
        int r = j + 8 * half, c = nt * 16 + nl;
        Pb[wave][r * 32 + c] = (bf16_t)sc[nt][j];
      }
    // Cross-lane DS RAW within the wave: drain DS counter + block compiler
    // reordering of the fragment loads below above the stores.
    asm volatile("s_wait_dscnt 0x0" ::: "memory");

    v16bf pa = frag_from_row(Pb[wave] + nl * 32, half);

    // P @ V for all 4 groups; V fragment rows are VtT[dim][key] -> b128 loads
#pragma unroll
    for (int g = 0; g < GRP; ++g) {
#pragma unroll
      for (int nt = 0; nt < 4; ++nt) {
        v16bf bv = frag_from_row(VtT[g] + (nt * 16 + nl) * 32, half);
        O[g][nt] = __builtin_amdgcn_wmma_f32_16x16x32_bf16(
            false, pa, false, bv, (short)0, O[g][nt], false, false);
      }
    }
  }

  // Normalize and write ctx (bf16) for all 4 full heads of this kv head
#pragma unroll
  for (int j = 0; j < 8; ++j) lrow[j] = 1.0f / lrow[j];
#pragma unroll
  for (int g = 0; g < GRP; ++g)
#pragma unroll
    for (int nt = 0; nt < 4; ++nt)
#pragma unroll
      for (int j = 0; j < 8; ++j) {
        int row = qbase + j + 8 * half;
        int col = (g * HKV + hk) * DK + nt * 16 + nl;
        Ctx[((size_t)(b * S_LEN + row)) * INNER + col] =
            (bf16_t)(O[g][nt][j] * lrow[j]);
      }
}

// ---------------------------------------------------------------------------
// Launcher
// ---------------------------------------------------------------------------
extern "C" void kernel_launch(void* const* d_in, const int* in_sizes, int n_in,
                              void* d_out, int out_size, void* d_ws, size_t ws_size,
                              hipStream_t stream) {
  const float* hs  = (const float*)d_in[0];  // [2,2048,1024]
  const float* Wq  = (const float*)d_in[1];  // [1024,1024]
  const float* Wk  = (const float*)d_in[2];
  const float* Wv  = (const float*)d_in[3];
  const float* Wo  = (const float*)d_in[4];
  const float* rel = (const float*)d_in[5];  // [32,16]
  float* out = (float*)d_out;                // [2,2048,1024]

  const int M = BATCH * S_LEN;               // 4096
  const size_t nHid = (size_t)M * DMODEL;    // 4M
  const size_t nW   = (size_t)DMODEL * INNER;// 1M

  char* ws = (char*)d_ws;
  size_t off = 0;
  auto alloc = [&](size_t bytes) {
    void* p = ws + off;
    off = (off + bytes + 255) & ~(size_t)255;
    return p;
  };
  bf16_t* hb    = (bf16_t*)alloc(nHid * 2);
  bf16_t* wq_b  = (bf16_t*)alloc(nW * 2);
  bf16_t* wk_b  = (bf16_t*)alloc(nW * 2);
  bf16_t* wv_b  = (bf16_t*)alloc(nW * 2);
  bf16_t* wo_b  = (bf16_t*)alloc(nW * 2);
  bf16_t* Qb    = (bf16_t*)alloc(nHid * 2);
  bf16_t* Kb    = (bf16_t*)alloc(nHid * 2);
  bf16_t* Vb    = (bf16_t*)alloc(nHid * 2);
  bf16_t* Qsum  = (bf16_t*)alloc((size_t)BATCH * HKV * S_LEN * DK * 2);
  bf16_t* Kmean = (bf16_t*)alloc((size_t)BATCH * HKV * S_LEN * DK * 2);
  bf16_t* ctx   = (bf16_t*)alloc(nHid * 2);
  float*  bmean = (float*)alloc(128 * 4);

  // 1) f32 -> bf16 conversions
  cvt_f32_bf16_kernel<<<(int)((nHid + 255) / 256), 256, 0, stream>>>(hs, hb, (int)nHid);
  cvt_f32_bf16_kernel<<<(int)((nW + 255) / 256), 256, 0, stream>>>(Wq, wq_b, (int)nW);
  cvt_f32_bf16_kernel<<<(int)((nW + 255) / 256), 256, 0, stream>>>(Wk, wk_b, (int)nW);
  cvt_f32_bf16_kernel<<<(int)((nW + 255) / 256), 256, 0, stream>>>(Wv, wv_b, (int)nW);
  cvt_f32_bf16_kernel<<<(int)((nW + 255) / 256), 256, 0, stream>>>(Wo, wo_b, (int)nW);

  // 2) Q/K/V projections (bf16 WMMA, bf16 out)
  dim3 gg(M / 128, INNER / 64);
  gemm_bf16_wmma_kernel<bf16_t><<<gg, 256, 0, stream>>>(hb, wq_b, Qb, M, INNER, DMODEL);
  gemm_bf16_wmma_kernel<bf16_t><<<gg, 256, 0, stream>>>(hb, wk_b, Kb, M, INNER, DMODEL);
  gemm_bf16_wmma_kernel<bf16_t><<<gg, 256, 0, stream>>>(hb, wv_b, Vb, M, INNER, DMODEL);

  // 3) Group reductions + bias table
  int nred = BATCH * HKV * S_LEN * DK;
  reduce_heads_kernel<<<(nred + 255) / 256, 256, 0, stream>>>(Qb, Kb, Qsum, Kmean);
  bias_prep_kernel<<<1, 128, 0, stream>>>(rel, bmean);

  // 4) Fused attention
  dim3 ga(S_LEN / 128, BATCH * HKV);
  flash_attn_kernel<<<ga, 256, 0, stream>>>(Qsum, Kmean, Vb, bmean, ctx);

  // 5) Output projection (f32 out)
  dim3 go(M / 128, DMODEL / 64);
  gemm_bf16_wmma_kernel<float><<<go, 256, 0, stream>>>(ctx, wo_b, out, M, DMODEL, INNER);
}
